// GNN_prompt_86646670229664
// MI455X (gfx1250) — matrix operand order repaired and checked by reading, not compiled, not run on hardware
//
#include <hip/hip_runtime.h>
#include <hip/hip_bf16.h>

#define N_NODES 100000
#define N_EDGES 1000000
#define DIM     128
#define DIM2    256
#define NLAYER  3
#define BN_EPS  1e-5f

typedef __attribute__((ext_vector_type(16))) __bf16 v16bf;
typedef __attribute__((ext_vector_type(8)))  float  v8f;

// ---------------------------------------------------------------------------
// h[i,c] = x_emb1[x1[i],c] + x_emb2[x2[i],c]; also pre-init layer-0
// aggregation buffer: agg = h + e_self(l=0).
// ---------------------------------------------------------------------------
__global__ void k_embed(const int* __restrict__ x1, const int* __restrict__ x2,
                        const float* __restrict__ e1, const float* __restrict__ e2,
                        const float* __restrict__ ee1, const float* __restrict__ ee2,
                        float* __restrict__ h, float* __restrict__ agg) {
    int idx = blockIdx.x * blockDim.x + threadIdx.x;
    if (idx >= N_NODES * DIM) return;
    int c = idx & (DIM - 1);
    int i = idx >> 7;
    float v = e1[x1[i] * DIM + c] + e2[x2[i] * DIM + c];
    h[idx]   = v;
    agg[idx] = v + ee1[4 * DIM + c] + ee2[c];   // self-loop attr (4,0)
}

// ---------------------------------------------------------------------------
// Per-layer prep: transpose weights to bf16 N-major (WMMA B-fragment loads
// become contiguous b128s), zero batch-norm stat accumulators.
//   w1bT[n*DIM  + k] = W1[k*DIM2 + n]   (k<128, n<256)
//   w2bT[n*DIM2 + k] = W2[k*DIM  + n]   (k<256, n<128)
// ---------------------------------------------------------------------------
__global__ void k_prep(const float* __restrict__ W1, const float* __restrict__ W2,
                       __bf16* __restrict__ w1bT, __bf16* __restrict__ w2bT,
                       float* __restrict__ stats) {
    int idx = blockIdx.x * blockDim.x + threadIdx.x;   // 0..32767
    if (idx < DIM * DIM2) {
        int k1 = idx / DIM2, n1 = idx % DIM2;
        w1bT[n1 * DIM + k1] = (__bf16)W1[idx];
        int k2 = idx / DIM, n2 = idx % DIM;
        w2bT[n2 * DIM2 + k2] = (__bf16)W2[idx];
    }
    if (idx < 2 * DIM) stats[idx] = 0.0f;
}

// ---------------------------------------------------------------------------
// Edge scatter-add: one wave32 per edge, lane owns 4 consecutive features.
// h and agg are L2-resident (51.2MB each, 192MB L2) -> gather + atomics hit
// L2, not HBM. Embedding tables (6x128 + 3x128 f32) staged in LDS.
// ---------------------------------------------------------------------------
__global__ void k_scatter(const int* __restrict__ src, const int* __restrict__ dst,
                          const int* __restrict__ ea1, const int* __restrict__ ea2,
                          const float* __restrict__ ee1, const float* __restrict__ ee2,
                          const float* __restrict__ h, float* __restrict__ agg) {
    __shared__ float s1[6 * DIM];
    __shared__ float s2[3 * DIM];
    for (int i = threadIdx.x; i < 6 * DIM; i += blockDim.x) s1[i] = ee1[i];
    for (int i = threadIdx.x; i < 3 * DIM; i += blockDim.x) s2[i] = ee2[i];
    __syncthreads();

    int lane  = threadIdx.x & 31;
    int wave  = (blockIdx.x * blockDim.x + threadIdx.x) >> 5;
    int nwave = (gridDim.x * blockDim.x) >> 5;
    int c = lane * 4;

    for (int e = wave; e < N_EDGES; e += nwave) {
        int s  = src[e];
        int d  = dst[e];
        int o1 = ea1[e] * DIM + c;
        int o2 = ea2[e] * DIM + c;
        float4 hv = *(const float4*)(h + s * DIM + c);
        float m0 = hv.x + s1[o1 + 0] + s2[o2 + 0];
        float m1 = hv.y + s1[o1 + 1] + s2[o2 + 1];
        float m2 = hv.z + s1[o1 + 2] + s2[o2 + 2];
        float m3 = hv.w + s1[o1 + 3] + s2[o2 + 3];
        float* p = agg + d * DIM + c;
        unsafeAtomicAdd(p + 0, m0);   // global_atomic_add_f32 @ L2
        unsafeAtomicAdd(p + 1, m1);
        unsafeAtomicAdd(p + 2, m2);
        unsafeAtomicAdd(p + 3, m3);
    }
}

// ---------------------------------------------------------------------------
// GEMM1: z1 = relu(agg[N,128] @ W1[128,256] + b1), stored bf16.
// Block = 4 waves, computes 16 rows x 256 cols. Each wave owns 4 adjacent
// 16x16 tiles (cols w*64 .. w*64+63) and REUSES one A fragment across 4
// WMMAs per k-step -> A read from L2 exactly once for the whole GEMM.
// A fragment (16-bit A 16x32, contiguous k order, not the ISA table order): lane m=lane&15, hi=lane>>4,
//   elems 0..15 -> K = hi*16+i.
// B fragment: lane column n, K = hi*16 + 0..15 (contiguous in N-major buf).
// ---------------------------------------------------------------------------
__global__ __launch_bounds__(128)
void k_gemm1(const float* __restrict__ A, const __bf16* __restrict__ WbT,
             const float* __restrict__ bias, __bf16* __restrict__ Zb) {
    int lane = threadIdx.x & 31;
    int wave = threadIdx.x >> 5;
    int row0 = blockIdx.x * 16;
    int colB = wave * 64;
    int m  = lane & 15;
    int hi = lane >> 4;

    const float* arow = A + (row0 + m) * DIM;
    const __bf16* bcol0 = WbT + (colB + m) * DIM;   // tile t adds t*16*DIM

    v8f c0 = {}, c1 = {}, c2 = {}, c3 = {};
    for (int kk = 0; kk < DIM; kk += 32) {
        v16bf a;
        int ka = kk + hi * 16;
#pragma unroll
        for (int i = 0; i < 8; ++i) {
            a[i]     = (__bf16)arow[ka + i];
            a[i + 8] = (__bf16)arow[ka + 8 + i];
        }
        int kb = kk + hi * 16;
        v16bf b0, b1, b2, b3;
#pragma unroll
        for (int i = 0; i < 16; ++i) {
            b0[i] = bcol0[kb + i];
            b1[i] = bcol0[16 * DIM + kb + i];
            b2[i] = bcol0[32 * DIM + kb + i];
            b3[i] = bcol0[48 * DIM + kb + i];
        }
        c0 = __builtin_amdgcn_wmma_f32_16x16x32_bf16(false, a, false, b0, (short)0, c0, false, false);
        c1 = __builtin_amdgcn_wmma_f32_16x16x32_bf16(false, a, false, b1, (short)0, c1, false, false);
        c2 = __builtin_amdgcn_wmma_f32_16x16x32_bf16(false, a, false, b2, (short)0, c2, false, false);
        c3 = __builtin_amdgcn_wmma_f32_16x16x32_bf16(false, a, false, b3, (short)0, c3, false, false);
    }

#pragma unroll
    for (int t = 0; t < 4; ++t) {
        v8f c = (t == 0) ? c0 : (t == 1) ? c1 : (t == 2) ? c2 : c3;
        int n = colB + t * 16 + m;
        float bv = bias[n];
#pragma unroll
        for (int r = 0; r < 8; ++r) {
            int mm = row0 + r + hi * 8;
            float v = c[r] + bv;
            v = v > 0.0f ? v : 0.0f;              // ReLU
            Zb[mm * DIM2 + n] = (__bf16)v;
        }
    }
}

// ---------------------------------------------------------------------------
// GEMM2: z = z1[N,256] @ W2[256,128] + b2 (fp32 out), fused per-column
// batch-norm sum/sumsq via fp32 atomics. Block = 4 waves = 16 rows x 128
// cols; each wave owns 2 tiles, reusing the A fragment.
// ---------------------------------------------------------------------------
__global__ __launch_bounds__(128)
void k_gemm2(const __bf16* __restrict__ Ab, const __bf16* __restrict__ WbT,
             const float* __restrict__ bias, float* __restrict__ Z,
             float* __restrict__ stats) {
    int lane = threadIdx.x & 31;
    int wave = threadIdx.x >> 5;
    int row0 = blockIdx.x * 16;
    int colB = wave * 32;
    int m  = lane & 15;
    int hi = lane >> 4;

    const __bf16* arow  = Ab  + (row0 + m) * DIM2;
    const __bf16* bcol0 = WbT + (colB + m) * DIM2;

    v8f c0 = {}, c1 = {};
    for (int kk = 0; kk < DIM2; kk += 32) {
        v16bf a;
        int ka = kk + hi * 16;
#pragma unroll
        for (int i = 0; i < 8; ++i) {
            a[i]     = arow[ka + i];
            a[i + 8] = arow[ka + 8 + i];
        }
        int kb = kk + hi * 16;
        v16bf b0, b1;
#pragma unroll
        for (int i = 0; i < 16; ++i) {
            b0[i] = bcol0[kb + i];
            b1[i] = bcol0[16 * DIM2 + kb + i];
        }
        c0 = __builtin_amdgcn_wmma_f32_16x16x32_bf16(false, a, false, b0, (short)0, c0, false, false);
        c1 = __builtin_amdgcn_wmma_f32_16x16x32_bf16(false, a, false, b1, (short)0, c1, false, false);
    }

#pragma unroll
    for (int t = 0; t < 2; ++t) {
        v8f c = (t == 0) ? c0 : c1;
        int n = colB + t * 16 + m;
        float bv = bias[n];
        float s = 0.0f, sq = 0.0f;
#pragma unroll
        for (int r = 0; r < 8; ++r) {
            int mm = row0 + r + hi * 8;
            float v = c[r] + bv;
            Z[mm * DIM + n] = v;
            s  += v;
            sq += v * v;
        }
        unsafeAtomicAdd(&stats[n], s);
        unsafeAtomicAdd(&stats[DIM + n], sq);
    }
}

// ---------------------------------------------------------------------------
// Finalize BN stats: stats[c] <- mean, stats[128+c] <- rsqrt(var + eps)
// ---------------------------------------------------------------------------
__global__ void k_bnfin(float* __restrict__ stats) {
    int c = threadIdx.x;
    if (c < DIM) {
        const float inv_n = 1.0f / (float)N_NODES;
        float mu  = stats[c] * inv_n;
        float var = stats[DIM + c] * inv_n - mu * mu;
        var = var > 0.0f ? var : 0.0f;
        stats[c]       = mu;
        stats[DIM + c] = rsqrtf(var + BN_EPS);
    }
}

// ---------------------------------------------------------------------------
// Apply BN (+ inter-layer ReLU), and pre-init the NEXT layer's agg buffer
// (agg_next = h + e_self_next) in the same pass when agg_next != nullptr.
// ---------------------------------------------------------------------------
__global__ void k_bnapply(const float* __restrict__ Z, const float* __restrict__ stats,
                          const float* __restrict__ gamma, const float* __restrict__ beta,
                          float* __restrict__ out, float* __restrict__ agg_next,
                          const float* __restrict__ ee1n, const float* __restrict__ ee2n,
                          int do_relu) {
    int idx = blockIdx.x * blockDim.x + threadIdx.x;
    if (idx >= N_NODES * DIM) return;
    int c = idx & (DIM - 1);
    float v = gamma[c] * (Z[idx] - stats[c]) * stats[DIM + c] + beta[c];
    if (do_relu) v = v > 0.0f ? v : 0.0f;
    out[idx] = v;
    if (agg_next) agg_next[idx] = v + ee1n[4 * DIM + c] + ee2n[c];
}

// ---------------------------------------------------------------------------
extern "C" void kernel_launch(void* const* d_in, const int* in_sizes, int n_in,
                              void* d_out, int out_size, void* d_ws, size_t ws_size,
                              hipStream_t stream) {
    const int*   x1        = (const int*)d_in[0];
    const int*   x2        = (const int*)d_in[1];
    const int*   edge_idx  = (const int*)d_in[2];
    const int*   ea1       = (const int*)d_in[3];
    const int*   ea2       = (const int*)d_in[4];
    const float* x_emb1    = (const float*)d_in[5];
    const float* x_emb2    = (const float*)d_in[6];
    const float* edge_emb1 = (const float*)d_in[7];
    const float* edge_emb2 = (const float*)d_in[8];
    const float* W1        = (const float*)d_in[9];
    const float* b1        = (const float*)d_in[10];
    const float* W2        = (const float*)d_in[11];
    const float* b2        = (const float*)d_in[12];
    const float* gamma     = (const float*)d_in[13];
    const float* beta      = (const float*)d_in[14];
    float* out = (float*)d_out;

    const int* src = edge_idx;
    const int* dst = edge_idx + N_EDGES;

    // workspace carve-up (256B aligned)
    char*  ws  = (char*)d_ws;
    size_t off = 0;
    auto carve = [&](size_t bytes) -> void* {
        void* p = ws + off;
        off = (off + bytes + 255) & ~(size_t)255;
        return p;
    };
    float*  hbuf  = (float*) carve((size_t)N_NODES * DIM * 4);   // node features
    float*  agg   = (float*) carve((size_t)N_NODES * DIM * 4);   // agg, reused as z2
    __bf16* z1b   = (__bf16*)carve((size_t)N_NODES * DIM2 * 2);  // hidden (bf16)
    __bf16* w1bT  = (__bf16*)carve((size_t)DIM * DIM2 * 2);      // W1^T bf16
    __bf16* w2bT  = (__bf16*)carve((size_t)DIM2 * DIM * 2);      // W2^T bf16
    float*  stats = (float*) carve(2 * DIM * 4);                 // sum/mu, sumsq/rsig

    const int ND = N_NODES * DIM;
    const int eltBlocks = (ND + 255) / 256;

    k_embed<<<eltBlocks, 256, 0, stream>>>(x1, x2, x_emb1, x_emb2,
                                           edge_emb1, edge_emb2, hbuf, agg);

    for (int l = 0; l < NLAYER; ++l) {
        const float* ee1 = edge_emb1 + l * 6 * DIM;
        const float* ee2 = edge_emb2 + l * 3 * DIM;

        k_prep<<<(DIM * DIM2 + 255) / 256, 256, 0, stream>>>(
            W1 + (size_t)l * DIM * DIM2, W2 + (size_t)l * DIM2 * DIM,
            w1bT, w2bT, stats);

        k_scatter<<<1024, 256, 0, stream>>>(src, dst, ea1, ea2, ee1, ee2, hbuf, agg);

        k_gemm1<<<N_NODES / 16, 128, 0, stream>>>(agg, w1bT, b1 + l * DIM2, z1b);

        k_gemm2<<<N_NODES / 16, 128, 0, stream>>>(z1b, w2bT, b2 + l * DIM, agg, stats);

        k_bnfin<<<1, 128, 0, stream>>>(stats);

        bool last = (l == NLAYER - 1);
        float* hout = last ? out : hbuf;
        float* aggn = last ? nullptr : agg;      // pre-init next layer's agg
        const float* ee1n = last ? edge_emb1 : edge_emb1 + (l + 1) * 6 * DIM;
        const float* ee2n = last ? edge_emb2 : edge_emb2 + (l + 1) * 3 * DIM;
        k_bnapply<<<eltBlocks, 256, 0, stream>>>(agg, stats, gamma + l * DIM,
                                                 beta + l * DIM, hout, aggn,
                                                 ee1n, ee2n, last ? 0 : 1);
    }
}
